// AnomalyAttention_29532195127323
// MI455X (gfx1250) — matrix operand order repaired and checked by reading, not compiled, not run on hardware
//
#include <hip/hip_runtime.h>
#include <hip/hip_bf16.h>
#include <math.h>

#define L_SEQ  2048
#define NHEAD  8
#define EDIM   64
#define DDIM   64
#define NBATCH 2
#define WAVES  4
#define RS     (NHEAD * EDIM)          // 512: row stride (elems) of [B,L,H,64]
#define TENS_ELEMS ((size_t)NBATCH * L_SEQ * RS)   // 2,097,152 per tensor

typedef __attribute__((ext_vector_type(16))) __bf16 v16bf;
typedef __attribute__((ext_vector_type(8)))  __bf16 v8bf;
typedef __attribute__((ext_vector_type(8)))  float  v8f;
typedef __attribute__((ext_vector_type(4)))  unsigned tdm_g0_t;
typedef __attribute__((ext_vector_type(8)))  int      tdm_g1_t;
typedef __attribute__((ext_vector_type(4)))  int      tdm_g2_t;

static __device__ __forceinline__ v8f wmma_bf16(v16bf a, v16bf b, v8f c) {
  return __builtin_amdgcn_wmma_f32_16x16x32_bf16(false, a, false, b, (short)0, c,
                                                 false, false);
}

// build a v16bf from two aligned 8-element LDS chunks (p[0..7], p[16..23])
static __device__ __forceinline__ v16bf ld16(const __bf16* p) {
  const v8bf a = *(const v8bf*)p;
  const v8bf b = *(const v8bf*)(p + 16);
  v16bf r;
#pragma unroll
  for (int i = 0; i < 8; ++i) { r[i] = a[i]; r[i + 8] = b[i]; }
  return r;
}

// ---- Tensor Data Mover: async 2D bf16 tile (global -> LDS), D# per ISA 08 ----
static __device__ __forceinline__ void tdm_load_2d(
    unsigned lds_off, const __bf16* gptr,
    unsigned tile_d0, unsigned tile_d1,
    unsigned tensor_d0, unsigned tensor_d1, unsigned stride_d0 /*elems*/)
{
  const unsigned long long ga = (unsigned long long)(uintptr_t)gptr;
  tdm_g0_t g0;
  g0[0] = 1u;                                         // count=1, user mode
  g0[1] = lds_off;                                    // LDS byte address
  g0[2] = (unsigned)(ga & 0xFFFFFFFFu);               // global_addr[31:0]
  g0[3] = (unsigned)((ga >> 32) & 0x01FFFFFFu) | (2u << 30);  // addr[56:32]|type=2
  tdm_g1_t g1;
  g1[0] = (int)(1u << 16);                            // data_size=1 (2 bytes)
  g1[1] = (int)((tensor_d0 & 0xFFFFu) << 16);         // tensor_dim0[15:0]
  g1[2] = (int)((tensor_d0 >> 16) | ((tensor_d1 & 0xFFFFu) << 16));
  g1[3] = (int)((tensor_d1 >> 16) | (tile_d0 << 16)); // tile_dim0
  g1[4] = (int)(tile_d1 & 0xFFFFu);                   // tile_dim1 (tile_dim2=0)
  g1[5] = (int)stride_d0;                             // tensor_dim0_stride lo32
  g1[6] = 0;                                          // stride hi | dim1_stride lo
  g1[7] = 0;
  tdm_g2_t z4 = {0, 0, 0, 0};
#if defined(__clang_major__) && __clang_major__ >= 23
  tdm_g1_t z8 = {0, 0, 0, 0, 0, 0, 0, 0};
  __builtin_amdgcn_tensor_load_to_lds(g0, g1, z4, z4, z8, 0);
#else
  __builtin_amdgcn_tensor_load_to_lds(g0, g1, z4, z4, 0);
#endif
}

// ================= pre-pass 1: elementwise f32 -> bf16 (K tensors) ==========
__global__ __launch_bounds__(256)
void cvt_bf16_kernel(const float* __restrict__ a, const float* __restrict__ b,
                     __bf16* __restrict__ oa, __bf16* __restrict__ ob)
{
  const size_t i = ((size_t)blockIdx.x * 256 + threadIdx.x) * 4;
  const float* s = blockIdx.y ? b : a;
  __bf16*      d = blockIdx.y ? ob : oa;
  const float4 v = *(const float4*)(s + i);
  d[i + 0] = (__bf16)v.x; d[i + 1] = (__bf16)v.y;
  d[i + 2] = (__bf16)v.z; d[i + 3] = (__bf16)v.w;
}

// ===== pre-pass 2: V [B,L,H,64] f32 -> Vt [B,H,64,L] bf16 (LDS transpose) ====
__global__ __launch_bounds__(256)
void vtrans_kernel(const float* __restrict__ vt, const float* __restrict__ vc,
                   __bf16* __restrict__ ot, __bf16* __restrict__ oc)
{
  __shared__ __bf16 t[64][65];
  const float* src = blockIdx.z ? vc : vt;
  __bf16*      dst = blockIdx.z ? oc : ot;
  const int b = blockIdx.y >> 3, h = blockIdx.y & 7;
  const int l0 = blockIdx.x * 64;
  const int tid = threadIdx.x;
  {
    const int lrow = tid >> 2, d0 = (tid & 3) * 16;
    const float* p = src + ((size_t)(b * L_SEQ + l0 + lrow) * NHEAD + h) * DDIM + d0;
#pragma unroll
    for (int i = 0; i < 16; ++i) t[lrow][d0 + i] = (__bf16)p[i];
  }
  __syncthreads();
  {
    const int drow = tid >> 2, k0 = (tid & 3) * 16;
    __bf16* q = dst + ((size_t)(b * NHEAD + h) * DDIM + drow) * L_SEQ + l0 + k0;
#pragma unroll
    for (int i = 0; i < 16; ++i) q[i] = t[k0 + i][drow];
  }
}

// ========================= main flash-attention kernel ======================
__global__ __launch_bounds__(WAVES * 32)
void anomaly_attn_fa(const float* __restrict__ qt, const float* __restrict__ qc,
                     const __bf16* __restrict__ kbf,   // [2][B,L,H,64] bf16
                     const __bf16* __restrict__ vbf,   // [2][B,H,64,L] bf16
                     float* __restrict__ out)
{
  __shared__ __attribute__((aligned(16))) __bf16 kbuf[2][64 * 64]; // [key][e]
  __shared__ __attribute__((aligned(16))) __bf16 vbuf[2][64 * 64]; // [d][key]
  __shared__ __attribute__((aligned(16))) __bf16 pbuf[WAVES][16 * 64];

  const int lane = threadIdx.x & 31;
  const int wave = threadIdx.x >> 5;
  const int half = lane >> 4;
  const int n    = lane & 15;

  const int b  = blockIdx.y / NHEAD;
  const int h  = blockIdx.y % NHEAD;
  const int z  = blockIdx.z;
  const int q0 = blockIdx.x * 64 + wave * 16;

  const float*  Q = (z ? qc : qt) + (size_t)b * L_SEQ * RS + (size_t)h * EDIM;
  const __bf16* K = kbf + (size_t)z * TENS_ELEMS
                        + (size_t)b * L_SEQ * RS + (size_t)h * EDIM;
  const __bf16* V = vbf + (size_t)z * TENS_ELEMS
                        + (size_t)(b * NHEAD + h) * DDIM * L_SEQ;
  float* O = out + (size_t)z * TENS_ELEMS
                 + (size_t)b * L_SEQ * RS + (size_t)h * EDIM;

  // ---- Q tile in A-layout, both 16x32 K-halves of E=64 ----
  v16bf aq0, aq1;
  {
    const float* qrow = Q + (size_t)(q0 + n) * RS;
#pragma unroll
    for (int i = 0; i < 16; ++i) {
      const int ka = ((i < 8) ? i : i + 8) + half * 8;
      aq0[i] = (__bf16)qrow[ka];
      aq1[i] = (__bf16)qrow[32 + ka];
    }
  }

  v8f o0 = {}, o1 = {}, o2 = {}, o3 = {};
  float rmax[8], rsum[8];
#pragma unroll
  for (int j = 0; j < 8; ++j) { rmax[j] = -INFINITY; rsum[j] = 0.0f; }
  const float scale = 0.125f;

  const int nch = blockIdx.x + 1;          // 64-key chunks up to the diagonal

  // prologue: async-stage chunk 0 (wave 0 drives the TDM)
  if (wave == 0) {
    tdm_load_2d((unsigned)(uintptr_t)&kbuf[0][0], K, 64, 64, 64, L_SEQ, RS);
    tdm_load_2d((unsigned)(uintptr_t)&vbuf[0][0], V, 64, 64, L_SEQ, 64, L_SEQ);
  }

  for (int ch = 0; ch < nch; ++ch) {
    const int c0 = ch * 64;
    __syncthreads();                        // prev users of buf[(ch+1)&1] done
    if (wave == 0) {
      if (ch + 1 < nch) {
        const int c1 = c0 + 64;
        tdm_load_2d((unsigned)(uintptr_t)&kbuf[(ch + 1) & 1][0],
                    K + (size_t)c1 * RS, 64, 64, 64, L_SEQ, RS);
        tdm_load_2d((unsigned)(uintptr_t)&vbuf[(ch + 1) & 1][0],
                    V + c1, 64, 64, L_SEQ, 64, L_SEQ);
        __builtin_amdgcn_s_wait_tensorcnt(2);   // chunk ch complete, ch+1 inflight
      } else {
        __builtin_amdgcn_s_wait_tensorcnt(0);
      }
    }
    __syncthreads();                        // chunk ch visible to all waves

    if (c0 <= q0 + 15) {                    // wave-uniform causal skip
      const __bf16* kb = kbuf[ch & 1];
      const __bf16* vb = vbuf[ch & 1];

      // ---- S = Q K^T : four 16-key subtiles ----
      v8f s[4];
#pragma unroll
      for (int sub = 0; sub < 4; ++sub) {
        const __bf16* kr = kb + (sub * 16 + n) * 64 + 8 * half;
        v8f acc = {};
        acc = wmma_bf16(aq0, ld16(kr), acc);
        acc = wmma_bf16(aq1, ld16(kr + 32), acc);
        s[sub] = acc;
      }

      // ---- online softmax ----
      const bool need_mask = (c0 + 63 > q0);   // only the diagonal chunk
      float p[4][8];
#pragma unroll
      for (int j = 0; j < 8; ++j) {
        const int qrow = q0 + j + 8 * half;
        float x[4];
#pragma unroll
        for (int sub = 0; sub < 4; ++sub) {
          x[sub] = s[sub][j] * scale;
          if (need_mask && (c0 + sub * 16 + n > qrow)) x[sub] = -INFINITY;
        }
        float mloc = fmaxf(fmaxf(x[0], x[1]), fmaxf(x[2], x[3]));
#pragma unroll
        for (int off = 1; off < 16; off <<= 1)
          mloc = fmaxf(mloc, __shfl_xor(mloc, off, 32));
        const float mnew = fmaxf(rmax[j], mloc);
        const float corr = __expf(rmax[j] - mnew);
        float ps = 0.0f;
#pragma unroll
        for (int sub = 0; sub < 4; ++sub) {
          const float e = __expf(x[sub] - mnew);
          p[sub][j] = e;
          ps += e;
        }
#pragma unroll
        for (int off = 1; off < 16; off <<= 1)
          ps += __shfl_xor(ps, off, 32);
        rsum[j] = rsum[j] * corr + ps;
        rmax[j] = mnew;
        o0[j] *= corr; o1[j] *= corr; o2[j] *= corr; o3[j] *= corr;
      }

      // ---- stage P (C-layout) into LDS, reload in A-layout ----
      __bf16* pw = pbuf[wave];
#pragma unroll
      for (int j = 0; j < 8; ++j)
#pragma unroll
        for (int sub = 0; sub < 4; ++sub)
          pw[(j + 8 * half) * 64 + sub * 16 + n] = (__bf16)p[sub][j];
      asm volatile("s_wait_dscnt 0" ::: "memory");
      v16bf ap0, ap1;
      {
        const __bf16* pr = pw + n * 64 + 8 * half;
        ap0 = ld16(pr);        // same values as rebuilt below
        // A-layout needs {k 8h..8h+7, 16+8h..23+8h} etc: assemble explicitly
        const v8bf a0 = *(const v8bf*)(pr);
        const v8bf a1 = *(const v8bf*)(pr + 16);
        const v8bf a2 = *(const v8bf*)(pr + 32);
        const v8bf a3 = *(const v8bf*)(pr + 48);
#pragma unroll
        for (int i = 0; i < 8; ++i) {
          ap0[i] = a0[i]; ap0[i + 8] = a1[i];
          ap1[i] = a2[i]; ap1[i + 8] = a3[i];
        }
      }

      // ---- O += P V : four 16-wide D tiles, K = 64 chunk keys ----
      v8f* oacc[4] = {&o0, &o1, &o2, &o3};
#pragma unroll
      for (int d = 0; d < 4; ++d) {
        const __bf16* vr = vb + (d * 16 + n) * 64 + 8 * half;
        v8f acc = *oacc[d];
        acc = wmma_bf16(ap0, ld16(vr), acc);
        acc = wmma_bf16(ap1, ld16(vr + 32), acc);
        *oacc[d] = acc;
      }
    }
  }

  // ---- normalize and store ----
#pragma unroll
  for (int j = 0; j < 8; ++j) {
    const float inv = 1.0f / rsum[j];
    float* op = O + (size_t)(q0 + j + 8 * half) * RS + n;
    op[0]  = o0[j] * inv;
    op[16] = o1[j] * inv;
    op[32] = o2[j] * inv;
    op[48] = o3[j] * inv;
  }
}

extern "C" void kernel_launch(void* const* d_in, const int* in_sizes, int n_in,
                              void* d_out, int out_size, void* d_ws, size_t ws_size,
                              hipStream_t stream) {
  const float* qt = (const float*)d_in[0];
  const float* kt = (const float*)d_in[1];
  const float* vt = (const float*)d_in[2];
  const float* qc = (const float*)d_in[3];
  const float* kc = (const float*)d_in[4];
  const float* vc = (const float*)d_in[5];

  __bf16* kbf = (__bf16*)d_ws;                       // [2][B,L,H,64]
  __bf16* vbf = kbf + 2 * TENS_ELEMS;                // [2][B,H,64,L]

  // pre-pass 1: K tensors -> bf16 (same layout)
  {
    dim3 g((unsigned)(TENS_ELEMS / 4 / 256), 2);
    cvt_bf16_kernel<<<g, 256, 0, stream>>>(kt, kc, kbf, kbf + TENS_ELEMS);
  }
  // pre-pass 2: V tensors -> bf16 transposed [B,H,64,L]
  {
    dim3 g(L_SEQ / 64, NBATCH * NHEAD, 2);
    vtrans_kernel<<<g, 256, 0, stream>>>(vt, vc, vbf, vbf + TENS_ELEMS);
  }
  // main fused dual attention
  {
    dim3 grid(L_SEQ / 64, NBATCH * NHEAD, 2);
    anomaly_attn_fa<<<grid, WAVES * 32, 0, stream>>>(qt, qc, kbf, vbf,
                                                     (float*)d_out);
  }
}
